// SuperpointGraph_6030134083771
// MI455X (gfx1250) — matrix operand order repaired and checked by reading, not compiled, not run on hardware
//
#include <hip/hip_runtime.h>
#include <hip/hip_bf16.h>
#include <cstdint>
#include <cstddef>

// ---------------------------------------------------------------------------
// Types for WMMA fragments (gfx1250, wave32)
// ---------------------------------------------------------------------------
typedef _Float16 half8 __attribute__((ext_vector_type(8)));
typedef _Float16 v16h  __attribute__((ext_vector_type(16)));
typedef float    v8f   __attribute__((ext_vector_type(8)));

#define NODES 1024   // B*N
#define NN    512    // N
#define CC    256    // C == OUT
#define EE    18

static __device__ __forceinline__ v8f wmma_f16(v16h a, v16h b, v8f c) {
  // D = A(16x32 f16) * B(32x16 f16) + C(16x16 f32)
  return __builtin_amdgcn_wmma_f32_16x16x32_f16(false, a, false, b, (short)0, c,
                                                false, false);
}

// B fragment via a lane-adjusted base pointer (bl = arena + lane*16).
// Fragment f lives at bl + f*512; two 16B loads.
static __device__ __forceinline__ v16h load_b_lane(const _Float16* bl, int frag) {
  const half8* p = (const half8*)(bl + (size_t)frag * 512);
  union { v16h v; half8 h[2]; } u;
  u.h[0] = p[0];
  u.h[1] = p[1];
  return u.v;
}

// A fragment from a row-major f16 tile (global or LDS).
// S points at row 0 of the 16-row tile; ld = row stride in halves.
// lane row = lane&15 ; halves: k = kbase + 8*(lane>>4) + {0..7, 16..23}
static __device__ __forceinline__ v16h load_a_rowmajor(const _Float16* S, int ld,
                                                       int kbase, int lane) {
  int r  = lane & 15;
  int kb = kbase + ((lane >> 4) << 3);
  const _Float16* p = S + (size_t)r * ld + kb;
  union { v16h v; half8 h[2]; } u;
  u.h[0] = *(const half8*)(p);
  u.h[1] = *(const half8*)(p + 16);
  return u.v;
}

// Store a D fragment (relu optional) into a row-major f16 staging tile.
// D layout: VGPR r -> row = r + 8*(lane>>4) ; col = lane&15 (+ n0)
static __device__ __forceinline__ void stage_relu_f16(_Float16* S, int ld, int n0,
                                                      v8f acc, int lane, bool relu) {
  int col = n0 + (lane & 15);
  int rb  = (lane >> 4) * 8;
#pragma unroll
  for (int r = 0; r < 8; ++r) {
    float v = acc[r];
    if (relu) v = fmaxf(v, 0.f);
    S[(size_t)(rb + r) * ld + col] = (_Float16)v;
  }
}

static __device__ __forceinline__ float wave_max(float v) {
#pragma unroll
  for (int o = 16; o; o >>= 1) v = fmaxf(v, __shfl_xor(v, o, 32));
  return v;
}
static __device__ __forceinline__ float wave_sum(float v) {
#pragma unroll
  for (int o = 16; o; o >>= 1) v += __shfl_xor(v, o, 32);
  return v;
}

// ---------------------------------------------------------------------------
// Prep: f32 -> f16 copy
// ---------------------------------------------------------------------------
__global__ void cvt16_kernel(const float* __restrict__ src, _Float16* __restrict__ dst, int n) {
  int i = blockIdx.x * 256 + threadIdx.x;
  if (i < n) dst[i] = (_Float16)src[i];
}

// ---------------------------------------------------------------------------
// Prep: pack a [Ksrc x Nf] row-major f32 weight (row stride lds) into the
// B-fragment arena layout, K padded to Kpad (multiple of 32) with zeros.
// Fragment (kt,nt): lane holds col nt*16+(lane&15); half h -> k = kt*32 +
// 8*(lane>>4) + (h<8 ? h : h+8).  Arena index = ((kt*NT+nt)*32 + lane)*16 + h.
// ---------------------------------------------------------------------------
__global__ void pack_b_kernel(const float* __restrict__ src, _Float16* __restrict__ dst,
                              int Kpad, int Ksrc, int Nf, int lds) {
  int idx = blockIdx.x * 256 + threadIdx.x;
  int total = Kpad * Nf;
  if (idx >= total) return;
  int frag = idx >> 9;          // 512 halves per fragment
  int wi   = idx & 511;
  int lane = wi >> 4;
  int h    = wi & 15;
  int NT = Nf >> 4;
  int kt = frag / NT, nt = frag % NT;
  int k = kt * 32 + ((lane >> 4) << 3) + (h < 8 ? h : h + 8);
  int n = nt * 16 + (lane & 15);
  float v = (k < Ksrc) ? src[(size_t)k * lds + n] : 0.f;
  dst[idx] = (_Float16)v;
}

// ---------------------------------------------------------------------------
// Generic wave-tile WMMA GEMM:  C[M,Nf] = act(A[M,K]_f16 @ Bfrag + bias)
// One 16x16 tile per wave. A row-major f16; B pre-swizzled fragment arena.
// Pointer-incremented K loop: constant-stride adds, immediate-offset b128s.
// ---------------------------------------------------------------------------
__global__ __launch_bounds__(128) void wmma_gemm_kernel(
    const _Float16* __restrict__ A, const _Float16* __restrict__ Bf,
    const float* __restrict__ bias,
    float* __restrict__ C32, int ld32,
    _Float16* __restrict__ C16, int ld16, int co16,
    int M, int K, int Nf, int relu) {
  int w = threadIdx.x >> 5, lane = threadIdx.x & 31;
  int NT = Nf >> 4, KT = K >> 5;
  int tiles = (M >> 4) * NT;
  int t = blockIdx.x * 4 + w;
  if (t >= tiles) return;               // wave-uniform: EXEC stays all-ones
  int mt = t / NT, nt = t % NT;
  int colL = lane & 15;
  int rb = (lane >> 4) * 8;

  float bv = bias ? bias[nt * 16 + colL] : 0.f;
  v8f acc = {bv, bv, bv, bv, bv, bv, bv, bv};

  const _Float16* ap = A + (size_t)(mt * 16 + colL) * K + ((lane >> 4) << 3);
  const _Float16* bp = Bf + ((size_t)nt * 32 + lane) * 16;
  const size_t bstep = (size_t)NT * 512;

  for (int kt = 0; kt < KT; ++kt) {
    union { v16h v; half8 h[2]; } ua, ub;
    ua.h[0] = *(const half8*)(ap);
    ua.h[1] = *(const half8*)(ap + 16);
    ub.h[0] = *(const half8*)(bp);
    ub.h[1] = *(const half8*)(bp + 8);
    acc = wmma_f16(ua.v, ub.v, acc);
    ap += 32;
    bp += bstep;
  }
  if (relu) {
#pragma unroll
    for (int r = 0; r < 8; ++r) acc[r] = fmaxf(acc[r], 0.f);
  }
  if (C32) {
    float* cp = C32 + (size_t)(mt * 16 + rb) * ld32 + nt * 16 + colL;
#pragma unroll
    for (int r = 0; r < 8; ++r) cp[(size_t)r * ld32] = acc[r];
  }
  if (C16) {
    _Float16* cp = C16 + (size_t)(mt * 16 + rb) * ld16 + co16 + nt * 16 + colL;
#pragma unroll
    for (int r = 0; r < 8; ++r) cp[(size_t)r * ld16] = (_Float16)acc[r];
  }
}

// ---------------------------------------------------------------------------
// Fused per-edge pipeline + online softmax message aggregation.
// One workgroup (8 wave32) per target node (b,i); each wave streams 4 tiles
// of 16 neighbor rows j through:
//   pe = relu-MLP(edge_feats)        (18->64->64->32)
//   h  = relu(pe@W + a_i + a_j + b); h2 = relu(h@W2+b2); score = h2.w3 + b3
//   online softmax over masked scores (flash-style)
//   g  = sigmoid(relu(pe@Weg + eg_x + b)@Weg2 + b2); msg += p * g * nbr_j
// Cross-wave softmax merge at the end; msg written as f16 into comb[:,256:].
//
// LDS: eg_w2 fragment arena (32KB, hottest weights: 32 of 60 WMMAs/tile) is
// copied in once per workgroup; per-wave staging is a single 16x64 f16 tile
// (pe/h2 overlay it after their A operands are in registers) + p + msg.
// ---------------------------------------------------------------------------
#define EG2_LDS_BYTES 32768
#define PWAVE   3136
#define OFF_S64 0          // 16x64 f16 staging (pe & h2 overlay rows 0..15 x 0..31)
#define OFF_SP  2048       // 16 f32 p-values
#define OFF_MSG 2112       // 256 f32 per-wave message
#define WSTATS_OFF (EG2_LDS_BYTES + 8 * PWAVE)          // 64 B
#define ATW3_OFF   (WSTATS_OFF + 64)                    // 128 B
#define SMEM_TOTAL (ATW3_OFF + 128)                     // 58048 B

// edge-weight arena offsets (halves)
#define EA_EM1  0
#define EA_EM2  2048
#define EA_EM3  6144
#define EA_AT1E 8192
#define EA_AT2  10240
#define EA_EG1E 12288
#define EA_EG2  14336

__global__ __launch_bounds__(256) void fused_edge_kernel(
    const float* __restrict__ edge_feats, const int* __restrict__ adj,
    const float* __restrict__ a_i_g, const float* __restrict__ a_j_g,
    const float* __restrict__ eg_x_g, const float* __restrict__ nbr_g,
    const _Float16* __restrict__ earena,
    const float* __restrict__ em_b1, const float* __restrict__ em_b2,
    const float* __restrict__ em_b3, const float* __restrict__ at_b1,
    const float* __restrict__ at_b2, const float* __restrict__ at_b3,
    const float* __restrict__ at_w3, const float* __restrict__ eg_b1,
    const float* __restrict__ eg_b2, _Float16* __restrict__ comb) {
  extern __shared__ char smem[];

  const int bi = blockIdx.x;            // b*N + i
  const int tid = threadIdx.x;
  const int w = tid >> 5, lane = tid & 31;
  const int colL = lane & 15;
  const int rb = (lane >> 4) * 8;
  const int base_node = bi & ~(NN - 1); // b*N

  // ---- one-time workgroup setup: eg_w2 arena + at_w3 into LDS -------------
  {
    const uint4* src = (const uint4*)(earena + EA_EG2);
    uint4* dst = (uint4*)smem;
#pragma unroll
    for (int it = 0; it < 8; ++it) dst[tid + 256 * it] = src[tid + 256 * it];
    if (tid < 32) ((float*)(smem + ATW3_OFF))[tid] = at_w3[tid];
  }

  char* wbase = smem + EG2_LDS_BYTES + (size_t)w * PWAVE;
  _Float16* S64 = (_Float16*)(wbase + OFF_S64);
  float*    Sp  = (float*)(wbase + OFF_SP);
  float*    msg = (float*)(wbase + OFF_MSG);
  float*    wstats = (float*)(smem + WSTATS_OFF);
  const float* w3s = (const float*)(smem + ATW3_OFF);

  // lane-adjusted fragment base pointers (frag f at +f*512 halves)
  const _Float16* bEM1  = earena + EA_EM1  + lane * 16;
  const _Float16* bEM2  = earena + EA_EM2  + lane * 16;
  const _Float16* bEM3  = earena + EA_EM3  + lane * 16;
  const _Float16* bAT1E = earena + EA_AT1E + lane * 16;
  const _Float16* bAT2  = earena + EA_AT2  + lane * 16;
  const _Float16* bEG1E = earena + EA_EG1E + lane * 16;
  const _Float16* bEG2L = (const _Float16*)smem + lane * 16;   // LDS-resident

#pragma unroll
  for (int r = 0; r < 8; ++r) msg[lane * 8 + r] = 0.f;
  float mrun = -3e38f, srun = 0.f;

  __syncthreads();   // eg2 arena + at_w3 visible to all waves

  for (int t = 0; t < 4; ++t) {
    const int jbase = w * 64 + t * 16;
    const int rowj  = jbase + colL;     // this lane's j for A-fragment rows
    const int jn0   = base_node + jbase + rb;   // first of this lane's 8 rows

    if (t < 3)
      __builtin_prefetch(edge_feats + ((size_t)bi * NN + (rowj + 16)) * EE, 0, 0);

    // ---- A fragment from edge_feats[b,i,j,:18], K padded to 32 -------------
    v16h Ae;
    {
      const float* ef = edge_feats + ((size_t)bi * NN + rowj) * EE;
      const int kb = (lane >> 4) * 8;   // 0 or 8
#pragma unroll
      for (int h = 0; h < 16; ++h) {
        int k = (h < 8) ? (kb + h) : (kb + 16 + (h - 8));
        float v = 0.f;
        if (h < 8) v = ef[k];                       // k <= 15 < 18: always valid
        else if (h < 10) { if (kb == 0) v = ef[k]; } // k = 16,17 valid; 24,25 not
        Ae[h] = (_Float16)v;                        // h >= 10: k >= 18 -> 0
      }
    }

    // ---- pe1 = relu(Ae @ em_w1 + b1)  [16,64] ------------------------------
#pragma unroll
    for (int nt = 0; nt < 4; ++nt) {
      float bv = em_b1[nt * 16 + colL];
      v8f acc = {bv, bv, bv, bv, bv, bv, bv, bv};
      acc = wmma_f16(Ae, load_b_lane(bEM1, nt), acc);
      stage_relu_f16(S64, 64, nt * 16, acc, lane, true);
    }
    __builtin_amdgcn_wave_barrier();

    // ---- pe2 = relu(pe1 @ em_w2 + b2)  [16,64] -----------------------------
    {
      v16h A0 = load_a_rowmajor(S64, 64, 0, lane);
      v16h A1 = load_a_rowmajor(S64, 64, 32, lane);
      __builtin_amdgcn_wave_barrier();
#pragma unroll
      for (int nt = 0; nt < 4; ++nt) {
        float bv = em_b2[nt * 16 + colL];
        v8f acc = {bv, bv, bv, bv, bv, bv, bv, bv};
        acc = wmma_f16(A0, load_b_lane(bEM2, nt), acc);
        acc = wmma_f16(A1, load_b_lane(bEM2, 4 + nt), acc);
        stage_relu_f16(S64, 64, nt * 16, acc, lane, true);
      }
    }
    __builtin_amdgcn_wave_barrier();

    // ---- pe = relu(pe2 @ em_w3 + b3)  [16,32] -> overlay S64[0:512] --------
    {
      v16h A0 = load_a_rowmajor(S64, 64, 0, lane);
      v16h A1 = load_a_rowmajor(S64, 64, 32, lane);
      __builtin_amdgcn_wave_barrier();
#pragma unroll
      for (int nt = 0; nt < 2; ++nt) {
        float bv = em_b3[nt * 16 + colL];
        v8f acc = {bv, bv, bv, bv, bv, bv, bv, bv};
        acc = wmma_f16(A0, load_b_lane(bEM3, nt), acc);
        acc = wmma_f16(A1, load_b_lane(bEM3, 2 + nt), acc);
        stage_relu_f16(S64, 32, nt * 16, acc, lane, true);
      }
    }
    __builtin_amdgcn_wave_barrier();

    v16h Ap = load_a_rowmajor(S64, 32, 0, lane);   // pe fragment, kept in VGPRs
    __builtin_amdgcn_wave_barrier();

    // ---- h = relu(pe @ at_w1_e + a_i + a_j + at_b1)  [16,64] ---------------
#pragma unroll
    for (int nt = 0; nt < 4; ++nt) {
      int c = nt * 16 + colL;
      float c0 = at_b1[c] + a_i_g[(size_t)bi * 64 + c];
      const float* ajp = a_j_g + (size_t)jn0 * 64 + c;
      v8f acc;
#pragma unroll
      for (int r = 0; r < 8; ++r) acc[r] = c0 + ajp[(size_t)r * 64];
      acc = wmma_f16(Ap, load_b_lane(bAT1E, nt), acc);
      stage_relu_f16(S64, 64, nt * 16, acc, lane, true);
    }
    __builtin_amdgcn_wave_barrier();

    // ---- h2 = relu(h @ at_w2 + b2)  [16,32] f16 -> overlay S64[0:512] ------
    {
      v16h A0 = load_a_rowmajor(S64, 64, 0, lane);
      v16h A1 = load_a_rowmajor(S64, 64, 32, lane);
      __builtin_amdgcn_wave_barrier();
#pragma unroll
      for (int nt = 0; nt < 2; ++nt) {
        float bv = at_b2[nt * 16 + colL];
        v8f acc = {bv, bv, bv, bv, bv, bv, bv, bv};
        acc = wmma_f16(A0, load_b_lane(bAT2, nt), acc);
        acc = wmma_f16(A1, load_b_lane(bAT2, 2 + nt), acc);
        stage_relu_f16(S64, 32, nt * 16, acc, lane, true);
      }
    }
    __builtin_amdgcn_wave_barrier();

    // ---- scores + masked online softmax ------------------------------------
    float score = -3e38f;
    if (lane < 16) {
      int j = jbase + lane;
      if (adj[(size_t)bi * NN + j] > 0) {
        float s = at_b3[0];
        const _Float16* hrow = S64 + lane * 32;
#pragma unroll
        for (int k = 0; k < 32; ++k) s += (float)hrow[k] * w3s[k];
        score = s;
      }
    }
    float tmax = wave_max(score);
    float newm = fmaxf(mrun, tmax);
    float p = (score > -1e30f) ? __expf(score - newm) : 0.f;
    float tsum = wave_sum(p);
    float scale = __expf(mrun - newm);
    srun = srun * scale + tsum;
    mrun = newm;
#pragma unroll
    for (int r = 0; r < 8; ++r) msg[lane * 8 + r] *= scale;
    if (lane < 16) Sp[lane] = p;
    __builtin_amdgcn_wave_barrier();

    // ---- g1 = relu(pe @ eg_w1_e + eg_x + b1)  [16,64] ----------------------
#pragma unroll
    for (int nt = 0; nt < 4; ++nt) {
      int c = nt * 16 + colL;
      float c0 = eg_b1[c];
      const float* exp_ = eg_x_g + (size_t)jn0 * 64 + c;
      v8f acc;
#pragma unroll
      for (int r = 0; r < 8; ++r) acc[r] = c0 + exp_[(size_t)r * 64];
      acc = wmma_f16(Ap, load_b_lane(bEG1E, nt), acc);
      stage_relu_f16(S64, 64, nt * 16, acc, lane, true);
    }
    __builtin_amdgcn_wave_barrier();

    // ---- g2 (LDS weights) -> sigmoid -> * nbr -> * p -> accumulate msg ------
    {
      v16h A0 = load_a_rowmajor(S64, 64, 0, lane);
      v16h A1 = load_a_rowmajor(S64, 64, 32, lane);
      float pv[8];
      {
        const float4* pp = (const float4*)(Sp + rb);
        float4 q0 = pp[0], q1 = pp[1];
        pv[0] = q0.x; pv[1] = q0.y; pv[2] = q0.z; pv[3] = q0.w;
        pv[4] = q1.x; pv[5] = q1.y; pv[6] = q1.z; pv[7] = q1.w;
      }
      __builtin_amdgcn_wave_barrier();
#pragma unroll
      for (int nt = 0; nt < 16; ++nt) {
        int c = nt * 16 + colL;
        float bv = eg_b2[c];
        v8f acc = {bv, bv, bv, bv, bv, bv, bv, bv};
        acc = wmma_f16(A0, load_b_lane(bEG2L, nt), acc);
        acc = wmma_f16(A1, load_b_lane(bEG2L, 16 + nt), acc);
        const float* nbp = nbr_g + (size_t)jn0 * CC + c;
        float colsum = 0.f;
#pragma unroll
        for (int r = 0; r < 8; ++r) {
          float g = 1.f / (1.f + __expf(-acc[r]));
          colsum += pv[r] * g * nbp[(size_t)r * CC];
        }
        colsum += __shfl_xor(colsum, 16, 32);
        if (lane < 16) msg[nt * 16 + lane] += colsum;
      }
    }
  }

  // ---- cross-wave softmax merge, write msg into comb[:, 256:512] -----------
  if (lane == 0) { wstats[2 * w] = mrun; wstats[2 * w + 1] = srun; }
  __syncthreads();

  float M = -3e38f;
#pragma unroll
  for (int ww = 0; ww < 8; ++ww) M = fmaxf(M, wstats[2 * ww]);
  float S = 0.f;
#pragma unroll
  for (int ww = 0; ww < 8; ++ww) S += wstats[2 * ww + 1] * __expf(wstats[2 * ww] - M);

  int c = tid;  // 0..255
  float acc = 0.f;
#pragma unroll
  for (int ww = 0; ww < 8; ++ww) {
    const float* mw = (const float*)(smem + EG2_LDS_BYTES + (size_t)ww * PWAVE + OFF_MSG);
    acc += mw[c] * __expf(wstats[2 * ww] - M);
  }
  float outv = (S > 0.f) ? acc / S : 0.f;   // has_nbr handling
  comb[(size_t)bi * 512 + 256 + c] = (_Float16)outv;
}

// ---------------------------------------------------------------------------
// Host launcher
// ---------------------------------------------------------------------------
extern "C" void kernel_launch(void* const* d_in, const int* in_sizes, int n_in,
                              void* d_out, int out_size, void* d_ws, size_t ws_size,
                              hipStream_t stream) {
  const float* x     = (const float*)d_in[0];
  const int*   adj   = (const int*)d_in[1];
  const float* ef    = (const float*)d_in[2];
  const float* st_w  = (const float*)d_in[3];
  const float* st_b  = (const float*)d_in[4];
  const float* nt_w  = (const float*)d_in[5];
  const float* nt_b  = (const float*)d_in[6];
  const float* em_w1 = (const float*)d_in[7];
  const float* em_b1 = (const float*)d_in[8];
  const float* em_w2 = (const float*)d_in[9];
  const float* em_b2 = (const float*)d_in[10];
  const float* em_w3 = (const float*)d_in[11];
  const float* em_b3 = (const float*)d_in[12];
  const float* at_w1 = (const float*)d_in[13];
  const float* at_b1 = (const float*)d_in[14];
  const float* at_w2 = (const float*)d_in[15];
  const float* at_b2 = (const float*)d_in[16];
  const float* at_w3 = (const float*)d_in[17];
  const float* at_b3 = (const float*)d_in[18];
  const float* eg_w1 = (const float*)d_in[19];
  const float* eg_b1 = (const float*)d_in[20];
  const float* eg_w2 = (const float*)d_in[21];
  const float* eg_b2 = (const float*)d_in[22];
  const float* cb_w1 = (const float*)d_in[23];
  const float* cb_b1 = (const float*)d_in[24];
  const float* cb_w2 = (const float*)d_in[25];
  const float* cb_b2 = (const float*)d_in[26];

  char* ws = (char*)d_ws;
  size_t off = 0;
  auto alloc = [&](size_t bytes) -> char* {
    char* p = ws + off;
    off = (off + bytes + 255) & ~(size_t)255;
    return p;
  };

  _Float16* x16       = (_Float16*)alloc((size_t)NODES * CC * 2);
  _Float16* arena_st  = (_Float16*)alloc((size_t)256 * 256 * 2);
  _Float16* arena_nt  = (_Float16*)alloc((size_t)256 * 256 * 2);
  _Float16* arena_ai  = (_Float16*)alloc((size_t)256 * 64 * 2);
  _Float16* arena_aj  = (_Float16*)alloc((size_t)256 * 64 * 2);
  _Float16* arena_egx = (_Float16*)alloc((size_t)256 * 64 * 2);
  _Float16* earena    = (_Float16*)alloc((size_t)30720 * 2);
  _Float16* arena_cb1 = (_Float16*)alloc((size_t)512 * 256 * 2);
  _Float16* arena_cb2 = (_Float16*)alloc((size_t)256 * 256 * 2);
  float*    nbr32     = (float*)alloc((size_t)NODES * CC * 4);
  float*    ai32      = (float*)alloc((size_t)NODES * 64 * 4);
  float*    aj32      = (float*)alloc((size_t)NODES * 64 * 4);
  float*    egx32     = (float*)alloc((size_t)NODES * 64 * 4);
  _Float16* comb      = (_Float16*)alloc((size_t)NODES * 512 * 2);
  _Float16* tmp16     = (_Float16*)alloc((size_t)NODES * 256 * 2);

  // ---- prep: x -> f16 ------------------------------------------------------
  {
    int n = NODES * CC;
    cvt16_kernel<<<(n + 255) / 256, 256, 0, stream>>>(x, x16, n);
  }

  // ---- prep: pack weights into B-fragment arenas ---------------------------
  auto pack = [&](const float* src, _Float16* dst, int Kpad, int Ksrc, int Nf, int lds) {
    int total = Kpad * Nf;
    pack_b_kernel<<<(total + 255) / 256, 256, 0, stream>>>(src, dst, Kpad, Ksrc, Nf, lds);
  };
  pack(st_w, arena_st, 256, 256, 256, 256);
  pack(nt_w, arena_nt, 256, 256, 256, 256);
  pack(at_w1, arena_ai, 256, 256, 64, 64);
  pack(at_w1 + 256 * 64, arena_aj, 256, 256, 64, 64);
  pack(eg_w1, arena_egx, 256, 256, 64, 64);
  pack(em_w1, earena + EA_EM1, 32, 18, 64, 64);
  pack(em_w2, earena + EA_EM2, 64, 64, 64, 64);
  pack(em_w3, earena + EA_EM3, 64, 64, 32, 32);
  pack(at_w1 + 512 * 64, earena + EA_AT1E, 32, 32, 64, 64);
  pack(at_w2, earena + EA_AT2, 64, 64, 32, 32);
  pack(eg_w1 + 256 * 64, earena + EA_EG1E, 32, 32, 64, 64);
  pack(eg_w2, earena + EA_EG2, 64, 64, 256, 256);
  pack(cb_w1, arena_cb1, 512, 512, 256, 256);
  pack(cb_w2, arena_cb2, 256, 256, 256, 256);

  // ---- node transforms (WMMA GEMMs) ----------------------------------------
  auto gemm = [&](const _Float16* A, const _Float16* Bf, const float* bias,
                  float* C32, int ld32, _Float16* C16, int ld16, int co16,
                  int M, int K, int Nf, int relu) {
    int tiles = (M / 16) * (Nf / 16);
    wmma_gemm_kernel<<<(tiles + 3) / 4, 128, 0, stream>>>(
        A, Bf, bias, C32, ld32, C16, ld16, co16, M, K, Nf, relu);
  };
  // self_feat -> comb[:, :256] (f16)
  gemm(x16, arena_st, st_b, nullptr, 0, comb, 512, 0, NODES, 256, 256, 0);
  // nbr (f32)
  gemm(x16, arena_nt, nt_b, nbr32, 256, nullptr, 0, 0, NODES, 256, 256, 0);
  // a_i, a_j, eg_x (no bias; biases applied in fused kernel)
  gemm(x16, arena_ai, nullptr, ai32, 64, nullptr, 0, 0, NODES, 256, 64, 0);
  gemm(x16, arena_aj, nullptr, aj32, 64, nullptr, 0, 0, NODES, 256, 64, 0);
  gemm(x16, arena_egx, nullptr, egx32, 64, nullptr, 0, 0, NODES, 256, 64, 0);

  // ---- fused per-edge pipeline ---------------------------------------------
  fused_edge_kernel<<<NODES, 256, SMEM_TOTAL, stream>>>(
      ef, adj, ai32, aj32, egx32, nbr32, earena,
      em_b1, em_b2, em_b3, at_b1, at_b2, at_b3, at_w3, eg_b1, eg_b2, comb);

  // ---- combine MLP: relu(comb @ cb_w1 + b1) @ cb_w2 + b2 -> out ------------
  gemm(comb, arena_cb1, cb_b1, nullptr, 0, tmp16, 256, 0, NODES, 512, 256, 1);
  gemm(tmp16, arena_cb2, cb_b2, (float*)d_out, 256, nullptr, 0, 0, NODES, 256, 256, 0);
}
